// AttnGCN3D_29334626632164
// MI455X (gfx1250) — matrix operand order repaired and checked by reading, not compiled, not run on hardware
//
#include <hip/hip_runtime.h>
#include <hip/hip_bf16.h>

// AttnGCN3D for MI455X (gfx1250): all dense layers via v_wmma_f32_16x16x32_f16.
// GEMM: wave computes 64x16 (4 M-tiles share one B fragment); weights pre-
// transposed to NxK so A and B fragments are pure 128-bit contiguous loads.
// Pipeline: topk(21) -> ndn -> surface conv -> 3x [WMMA gemm + gather/max combine]
// -> fc1/q/k/v gemms -> pos MLP (d1 VALU + d2 WMMA) -> u -> g1/g2 WMMA
// -> softmax over 16 neighbors -> fc2 WMMA with residual + transposed store.
// Workspace requirement ~310 MB.

#define NPTS 4096
#define BATCH 4
#define BN (BATCH * NPTS)
#define KNBR 20
#define ATTNK 16
#define COUT 128
#define INV_SQRT_D 0.08838834764831845f
#define FBIG 3.4e38f

typedef _Float16 h16;
typedef _Float16 v16h __attribute__((ext_vector_type(16)));
typedef _Float16 h8v  __attribute__((ext_vector_type(8)));
typedef float    v8f  __attribute__((ext_vector_type(8)));

// ---------------------------------------------------------------------------
// f32 -> f16 with transpose: dst[N x K] = src[K x N]^T   (weight prep)
// ---------------------------------------------------------------------------
__global__ void f2h_tr_kernel(const float* __restrict__ src, h16* __restrict__ dst,
                              int K, int N) {
    int i = blockIdx.x * 256 + threadIdx.x;
    if (i < K * N) {
        int k = i / N, n = i - k * N;
        dst[(size_t)n * K + k] = (h16)src[i];
    }
}

// ---------------------------------------------------------------------------
// Top-21 nearest neighbors per point (entry 0 = self).
// One block per point; 128 threads; full distance row staged in LDS, then 21
// min-extractions with tree reduction. Tie-break: lower index (matches top_k).
// ---------------------------------------------------------------------------
__global__ __launch_bounds__(128) void topk_kernel(const float* __restrict__ xyz,
                                                   int* __restrict__ idx21) {
    __shared__ float sd[NPTS];
    __shared__ float rv[128];
    __shared__ int   ri[128];
    int bn = blockIdx.x;
    int b = bn / NPTS, i = bn % NPTS;
    const float* xb = xyz + (size_t)b * 3 * NPTS;
    float xi = xb[i], yi = xb[NPTS + i], zi = xb[2 * NPTS + i];
    float sqi = xi * xi + yi * yi + zi * zi;
    int tid = threadIdx.x;
    for (int j = tid; j < NPTS; j += 128) {
        float xj = xb[j], yj = xb[NPTS + j], zj = xb[2 * NPTS + j];
        float sqj = xj * xj + yj * yj + zj * zj;
        sd[j] = sqi + sqj - 2.f * (xi * xj + yi * yj + zi * zj);
    }
    __syncthreads();
    const int chunk = NPTS / 128;   // 32
    int base = tid * chunk;
    for (int r = 0; r < KNBR + 1; ++r) {
        float bv = FBIG; int bi = 0x7fffffff;
        #pragma unroll 4
        for (int j = 0; j < chunk; ++j) {
            float v = sd[base + j];
            if (v < bv) { bv = v; bi = base + j; }
        }
        rv[tid] = bv; ri[tid] = bi;
        __syncthreads();
        for (int s = 64; s > 0; s >>= 1) {
            if (tid < s) {
                float ov = rv[tid + s]; int oi = ri[tid + s];
                if (ov < rv[tid] || (ov == rv[tid] && oi < ri[tid])) { rv[tid] = ov; ri[tid] = oi; }
            }
            __syncthreads();
        }
        if (tid == 0) { idx21[(size_t)bn * 21 + r] = ri[0]; sd[ri[0]] = FBIG; }
        __syncthreads();
    }
}

// ---------------------------------------------------------------------------
// Unit neighbor directions ndn (B,N,20,3)
// ---------------------------------------------------------------------------
__global__ __launch_bounds__(32) void ndn_kernel(const float* __restrict__ xyz,
                                                 const int* __restrict__ idx21,
                                                 float* __restrict__ ndn) {
    int bn = blockIdx.x;
    int k = threadIdx.x;
    if (k >= KNBR) return;
    int b = bn / NPTS, i = bn % NPTS;
    const float* xb = xyz + (size_t)b * 3 * NPTS;
    int j = idx21[(size_t)bn * 21 + 1 + k];
    float dx = xb[j] - xb[i];
    float dy = xb[NPTS + j] - xb[NPTS + i];
    float dz = xb[2 * NPTS + j] - xb[2 * NPTS + i];
    float nrm = fmaxf(sqrtf(dx * dx + dy * dy + dz * dz), 1e-12f);
    float inv = 1.f / nrm;
    float* o = ndn + ((size_t)bn * KNBR + k) * 3;
    o[0] = dx * inv; o[1] = dy * inv; o[2] = dz * inv;
}

// ---------------------------------------------------------------------------
// conv_surface: f0 = max_k relu(ndn_k . normalize(dirs col)), written f16
// ---------------------------------------------------------------------------
__global__ __launch_bounds__(128) void surf_kernel(const float* __restrict__ ndn,
                                                   const float* __restrict__ dirs,
                                                   h16* __restrict__ f0h) {
    __shared__ float snd[KNBR * 3];
    int bn = blockIdx.x, c = threadIdx.x;
    if (c < KNBR * 3) snd[c] = ndn[(size_t)bn * (KNBR * 3) + c];
    __syncthreads();
    float d0 = dirs[c], d1 = dirs[COUT + c], d2 = dirs[2 * COUT + c];
    float inv = 1.f / fmaxf(sqrtf(d0 * d0 + d1 * d1 + d2 * d2), 1e-12f);
    d0 *= inv; d1 *= inv; d2 *= inv;
    float m = 0.f;
    #pragma unroll
    for (int k = 0; k < KNBR; ++k) {
        float t = fmaxf(snd[k * 3] * d0 + snd[k * 3 + 1] * d1 + snd[k * 3 + 2] * d2, 0.f);
        m = fmaxf(m, t);
    }
    f0h[(size_t)bn * COUT + c] = (h16)m;   // relu(max of relu) == max
}

// ---------------------------------------------------------------------------
// WMMA GEMM: C[M,N] = A[M,K](f16) * B[K,N] + bias, opt relu.
// Bt is the weight pre-transposed to N x K. Each wave computes a 64x16 strip:
// 4 M-tiles share one B fragment per 32-deep k-step -> 4 v_wmma per step.
// A/B fragments are assembled from contiguous 128-bit loads.
// Epilogues: f32 store / f16 store / residual + transposed (B,C,N) store.
// ---------------------------------------------------------------------------
__global__ __launch_bounds__(256) void gemm64_wmma(
    const h16* __restrict__ A, const h16* __restrict__ Bt,
    const float* __restrict__ bias,
    float* __restrict__ outF, h16* __restrict__ outH,
    const float* __restrict__ resid, float* __restrict__ outT,
    int M, int N, int K, int relu) {
    int wave = threadIdx.x >> 5;
    int lane = threadIdx.x & 31;
    int ct = (blockIdx.y << 7) + (wave << 4);
    if (ct >= N) return;
    int lm = lane & 15;
    int hh = lane >> 4;
    int col = ct + lm;
    int mbase = blockIdx.x << 6;

    const h8v* Brow = (const h8v*)(Bt + (size_t)col * K);   // 16B-aligned (K mult of 8)
    const h8v* Ar0 = (const h8v*)(A + (size_t)(mbase + lm) * K);
    const h8v* Ar1 = (const h8v*)(A + (size_t)(mbase + 16 + lm) * K);
    const h8v* Ar2 = (const h8v*)(A + (size_t)(mbase + 32 + lm) * K);
    const h8v* Ar3 = (const h8v*)(A + (size_t)(mbase + 48 + lm) * K);

    v8f acc[4];
    #pragma unroll
    for (int t = 0; t < 4; ++t) acc[t] = (v8f){0.f, 0.f, 0.f, 0.f, 0.f, 0.f, 0.f, 0.f};

    for (int kk = 0; kk < K; kk += 32) {
        // B fragment: lane(col) holds K = kk+16*hh .. +15 -> 16 contiguous halves
        int ib = (kk + 16 * hh) >> 3;
        h8v b0 = Brow[ib], b1 = Brow[ib + 1];
        v16h b = __builtin_shufflevector(b0, b1, 0, 1, 2, 3, 4, 5, 6, 7,
                                                 8, 9, 10, 11, 12, 13, 14, 15);
        // A fragment: halves [kk+16h .. +15] contiguous (matches B)
        int i0 = (kk + 16 * hh) >> 3;
        int i1 = i0 + 1;
        #pragma unroll
        for (int t = 0; t < 4; ++t) {
            const h8v* Ar = (t == 0) ? Ar0 : (t == 1) ? Ar1 : (t == 2) ? Ar2 : Ar3;
            h8v a0 = Ar[i0], a1 = Ar[i1];
            v16h a = __builtin_shufflevector(a0, a1, 0, 1, 2, 3, 4, 5, 6, 7,
                                                     8, 9, 10, 11, 12, 13, 14, 15);
            acc[t] = __builtin_amdgcn_wmma_f32_16x16x32_f16(false, a, false, b,
                                                            (short)0, acc[t], false, false);
        }
    }

    float bcol = bias ? bias[col] : 0.f;
    if (outT) {
        #pragma unroll
        for (int t = 0; t < 4; ++t) {
            #pragma unroll
            for (int r = 0; r < 8; ++r) {
                int m = mbase + 16 * t + r + 8 * hh;
                float v = acc[t][r] + bcol;
                size_t o = (size_t)m * N + col;
                if (resid) v += resid[o];
                int bb = m / NPTS, nn = m & (NPTS - 1);
                outT[((size_t)bb * N + col) * NPTS + nn] = v;
            }
        }
    } else {
        #pragma unroll
        for (int t = 0; t < 4; ++t) {
            #pragma unroll
            for (int r = 0; r < 8; ++r) {
                int m = mbase + 16 * t + r + 8 * hh;
                float v = acc[t][r] + bcol;
                if (relu) v = fmaxf(v, 0.f);
                size_t o = (size_t)m * N + col;
                if (outF) outF[o] = v;
                if (outH) outH[o] = (h16)v;
            }
        }
    }
}

// ---------------------------------------------------------------------------
// conv_layer combine: out = relu(fo[:,c] + max_k relu(ndn_k.dir_c) * fo[nbr_k, oc+c])
// ---------------------------------------------------------------------------
__global__ __launch_bounds__(256) void conv_combine_kernel(
    const float* __restrict__ fo, const int* __restrict__ idx21,
    const float* __restrict__ ndn, const float* __restrict__ dirs,
    h16* __restrict__ outH, float* __restrict__ outF, int oc) {
    __shared__ float snd[KNBR * 3];
    __shared__ int   snb[KNBR];
    int bn = blockIdx.x, c = threadIdx.x;
    int b = bn / NPTS;
    if (c < KNBR * 3) snd[c] = ndn[(size_t)bn * (KNBR * 3) + c];
    if (c < KNBR)     snb[c] = idx21[(size_t)bn * 21 + 1 + c];
    __syncthreads();
    float d0 = dirs[c], d1 = dirs[oc + c], d2 = dirs[2 * oc + c];
    float inv = 1.f / fmaxf(sqrtf(d0 * d0 + d1 * d1 + d2 * d2), 1e-12f);
    d0 *= inv; d1 *= inv; d2 *= inv;
    int stride = 2 * oc;
    float m = -FBIG;
    #pragma unroll 4
    for (int k = 0; k < KNBR; ++k) {
        float th = fmaxf(snd[k * 3] * d0 + snd[k * 3 + 1] * d1 + snd[k * 3 + 2] * d2, 0.f);
        int row = b * NPTS + snb[k];
        float sup = fo[(size_t)row * stride + oc + c];
        m = fmaxf(m, th * sup);
    }
    float v = fmaxf(fo[(size_t)bn * stride + c] + m, 0.f);
    outH[(size_t)bn * oc + c] = (h16)v;
    if (outF) outF[(size_t)bn * oc + c] = v;
}

// ---------------------------------------------------------------------------
// pos head: t[bn,k,c] = relu((verts_i - verts_knn_k) . d1_w[:,c] + d1_b[c])  (f16)
// ---------------------------------------------------------------------------
__global__ __launch_bounds__(128) void poshead_kernel(const float* __restrict__ xyz,
                                                      const int* __restrict__ idx21,
                                                      const float* __restrict__ d1w,
                                                      const float* __restrict__ d1b,
                                                      h16* __restrict__ tH) {
    __shared__ float rel[ATTNK * 3];
    int bn = blockIdx.x, c = threadIdx.x;
    int b = bn / NPTS, i = bn % NPTS;
    const float* xb = xyz + (size_t)b * 3 * NPTS;
    if (c < ATTNK) {
        int j = idx21[(size_t)bn * 21 + c];   // knn includes self (first 16 of top-21)
        rel[c * 3 + 0] = xb[i] - xb[j];
        rel[c * 3 + 1] = xb[NPTS + i] - xb[NPTS + j];
        rel[c * 3 + 2] = xb[2 * NPTS + i] - xb[2 * NPTS + j];
    }
    __syncthreads();
    float w0 = d1w[c], w1 = d1w[COUT + c], w2 = d1w[2 * COUT + c], bb = d1b[c];
    #pragma unroll
    for (int k = 0; k < ATTNK; ++k) {
        float v = fmaxf(rel[k * 3] * w0 + rel[k * 3 + 1] * w1 + rel[k * 3 + 2] * w2 + bb, 0.f);
        tH[((size_t)bn * ATTNK + k) * COUT + c] = (h16)v;
    }
}

// ---------------------------------------------------------------------------
// u[bn,k,c] = q[bn,c] - kx[knn_k, c] + pos[bn,k,c]   (f16)
// ---------------------------------------------------------------------------
__global__ __launch_bounds__(128) void u_kernel(const float* __restrict__ q,
                                                const float* __restrict__ kx,
                                                const h16* __restrict__ posH,
                                                const int* __restrict__ idx21,
                                                h16* __restrict__ uH) {
    __shared__ int sk[ATTNK];
    int bn = blockIdx.x, c = threadIdx.x;
    int b = bn / NPTS;
    if (c < ATTNK) sk[c] = idx21[(size_t)bn * 21 + c];
    __syncthreads();
    float qv = q[(size_t)bn * COUT + c];
    #pragma unroll
    for (int k = 0; k < ATTNK; ++k) {
        int row = b * NPTS + sk[k];
        float v = qv - kx[(size_t)row * COUT + c] + (float)posH[((size_t)bn * ATTNK + k) * COUT + c];
        uH[((size_t)bn * ATTNK + k) * COUT + c] = (h16)v;
    }
}

// ---------------------------------------------------------------------------
// Per-feature softmax over 16 neighbors + weighted sum of (v + pos) -> res (f16)
// ---------------------------------------------------------------------------
__global__ __launch_bounds__(128) void attn_kernel(const h16* __restrict__ logitsH,
                                                   const float* __restrict__ vx,
                                                   const h16* __restrict__ posH,
                                                   const int* __restrict__ idx21,
                                                   h16* __restrict__ resH) {
    __shared__ int sk[ATTNK];
    int bn = blockIdx.x, c = threadIdx.x;
    int b = bn / NPTS;
    if (c < ATTNK) sk[c] = idx21[(size_t)bn * 21 + c];
    __syncthreads();
    float l[ATTNK];
    float mx = -FBIG;
    #pragma unroll
    for (int k = 0; k < ATTNK; ++k) {
        l[k] = (float)logitsH[((size_t)bn * ATTNK + k) * COUT + c] * INV_SQRT_D;
        mx = fmaxf(mx, l[k]);
    }
    float s = 0.f;
    #pragma unroll
    for (int k = 0; k < ATTNK; ++k) { l[k] = __expf(l[k] - mx); s += l[k]; }
    float inv = 1.f / s;
    float acc = 0.f;
    #pragma unroll
    for (int k = 0; k < ATTNK; ++k) {
        int row = b * NPTS + sk[k];
        float val = vx[(size_t)row * COUT + c] + (float)posH[((size_t)bn * ATTNK + k) * COUT + c];
        acc += l[k] * inv * val;
    }
    resH[(size_t)bn * COUT + c] = (h16)acc;
}

// ---------------------------------------------------------------------------
// Host launch
// ---------------------------------------------------------------------------
extern "C" void kernel_launch(void* const* d_in, const int* in_sizes, int n_in,
                              void* d_out, int out_size, void* d_ws, size_t ws_size,
                              hipStream_t stream) {
    const float* xyz        = (const float*)d_in[0];
    const float* conv0_dirs = (const float*)d_in[1];
    const float* conv1_w    = (const float*)d_in[2];
    const float* conv1_b    = (const float*)d_in[3];
    const float* conv1_dirs = (const float*)d_in[4];
    const float* conv2_w    = (const float*)d_in[5];
    const float* conv2_b    = (const float*)d_in[6];
    const float* conv2_dirs = (const float*)d_in[7];
    const float* conv3_w    = (const float*)d_in[8];
    const float* conv3_b    = (const float*)d_in[9];
    const float* conv3_dirs = (const float*)d_in[10];
    const float* fc1_w = (const float*)d_in[11]; const float* fc1_b = (const float*)d_in[12];
    const float* fc2_w = (const float*)d_in[13]; const float* fc2_b = (const float*)d_in[14];
    const float* d1_w  = (const float*)d_in[15]; const float* d1_b  = (const float*)d_in[16];
    const float* d2_w  = (const float*)d_in[17]; const float* d2_b  = (const float*)d_in[18];
    const float* g1_w  = (const float*)d_in[19]; const float* g1_b  = (const float*)d_in[20];
    const float* g2_w  = (const float*)d_in[21]; const float* g2_b  = (const float*)d_in[22];
    const float* wq    = (const float*)d_in[23];
    const float* wk    = (const float*)d_in[24];
    const float* wv    = (const float*)d_in[25];
    float* out = (float*)d_out;

    // ---- bump workspace allocator ----
    char* wsb = (char*)d_ws;
    size_t off = 0;
    auto alloc = [&](size_t bytes) -> void* {
        void* p = wsb + off;
        off = (off + bytes + 255) & ~(size_t)255;
        return p;
    };
    const size_t BN16 = (size_t)BN * ATTNK;

    int*   idx21 = (int*)  alloc((size_t)BN * 21 * 4);
    float* ndn   = (float*)alloc((size_t)BN * KNBR * 3 * 4);
    // f16 weights, transposed to N x K
    h16* w1h  = (h16*)alloc(128 * 256 * 2);
    h16* w2h  = (h16*)alloc(128 * 512 * 2);
    h16* w3h  = (h16*)alloc(256 * 256 * 2);
    h16* fc1h = (h16*)alloc(128 * 128 * 2);
    h16* fc2h = (h16*)alloc(128 * 128 * 2);
    h16* d2h  = (h16*)alloc(128 * 128 * 2);
    h16* g1h  = (h16*)alloc(128 * 128 * 2);
    h16* g2h  = (h16*)alloc(128 * 128 * 2);
    h16* wqh  = (h16*)alloc(128 * 128 * 2);
    h16* wkh  = (h16*)alloc(128 * 128 * 2);
    h16* wvh  = (h16*)alloc(128 * 128 * 2);
    // activations
    h16*   f0h = (h16*)  alloc((size_t)BN * 128 * 2);
    h16*   f1h = (h16*)  alloc((size_t)BN * 128 * 2);
    h16*   f2h = (h16*)  alloc((size_t)BN * 256 * 2);
    float* f3f = (float*)alloc((size_t)BN * 128 * 4);
    h16*   f3h = (h16*)  alloc((size_t)BN * 128 * 2);
    h16*   xh  = (h16*)  alloc((size_t)BN * 128 * 2);
    float* qf  = (float*)alloc((size_t)BN * 128 * 4);
    float* kxf = (float*)alloc((size_t)BN * 128 * 4);
    float* vxf = (float*)alloc((size_t)BN * 128 * 4);
    float* fo  = (float*)alloc((size_t)BN * 512 * 4);        // reused conv1/2/3
    h16*   bigA = (h16*) alloc(BN16 * COUT * 2);             // t -> u -> logits
    h16*   posH = (h16*) alloc(BN16 * COUT * 2);
    h16*   a1H  = (h16*) alloc(BN16 * COUT * 2);
    h16*   resH = (h16*) alloc((size_t)BN * 128 * 2);
    if (off > ws_size) return;   // workspace too small (~310 MB required)

    auto cvt = [&](const float* s, h16* d, int K, int N) {
        int n = K * N;
        f2h_tr_kernel<<<dim3((n + 255) / 256), dim3(256), 0, stream>>>(s, d, K, N);
    };
    cvt(conv1_w, w1h, 128, 256); cvt(conv2_w, w2h, 128, 512); cvt(conv3_w, w3h, 256, 256);
    cvt(fc1_w, fc1h, 128, 128); cvt(fc2_w, fc2h, 128, 128); cvt(d2_w, d2h, 128, 128);
    cvt(g1_w, g1h, 128, 128);   cvt(g2_w, g2h, 128, 128);
    cvt(wq, wqh, 128, 128); cvt(wk, wkh, 128, 128); cvt(wv, wvh, 128, 128);

    auto gemm = [&](const h16* A, const h16* Bt, const float* bias,
                    float* oF, h16* oH, const float* resid, float* oT,
                    int M, int N, int K, int relu) {
        dim3 g(M / 64, (N + 127) / 128);
        gemm64_wmma<<<g, dim3(256), 0, stream>>>(A, Bt, bias, oF, oH, resid, oT, M, N, K, relu);
    };

    // neighbor graph
    topk_kernel<<<dim3(BN), dim3(128), 0, stream>>>(xyz, idx21);
    ndn_kernel<<<dim3(BN), dim3(32), 0, stream>>>(xyz, idx21, ndn);

    // conv stack
    surf_kernel<<<dim3(BN), dim3(128), 0, stream>>>(ndn, conv0_dirs, f0h);
    gemm(f0h, w1h, conv1_b, fo, nullptr, nullptr, nullptr, BN, 256, 128, 0);
    conv_combine_kernel<<<dim3(BN), dim3(128), 0, stream>>>(fo, idx21, ndn, conv1_dirs, f1h, nullptr, 128);
    gemm(f1h, w2h, conv2_b, fo, nullptr, nullptr, nullptr, BN, 512, 128, 0);
    conv_combine_kernel<<<dim3(BN), dim3(256), 0, stream>>>(fo, idx21, ndn, conv2_dirs, f2h, nullptr, 256);
    gemm(f2h, w3h, conv3_b, fo, nullptr, nullptr, nullptr, BN, 256, 256, 0);
    conv_combine_kernel<<<dim3(BN), dim3(128), 0, stream>>>(fo, idx21, ndn, conv3_dirs, f3h, f3f, 128);

    // transformer projections
    gemm(f3h, fc1h, fc1_b, nullptr, xh, nullptr, nullptr, BN, 128, 128, 0);
    gemm(xh, wqh, nullptr, qf,  nullptr, nullptr, nullptr, BN, 128, 128, 0);
    gemm(xh, wkh, nullptr, kxf, nullptr, nullptr, nullptr, BN, 128, 128, 0);
    gemm(xh, wvh, nullptr, vxf, nullptr, nullptr, nullptr, BN, 128, 128, 0);

    // positional MLP: t (VALU) then d2 (WMMA, M=262144)
    poshead_kernel<<<dim3(BN), dim3(128), 0, stream>>>(xyz, idx21, d1_w, d1_b, bigA);
    gemm(bigA, d2h, d2_b, nullptr, posH, nullptr, nullptr, (int)BN16, 128, 128, 0);

    // attention logits: u -> g1 -> g2  (bigA reused for u, then for logits)
    u_kernel<<<dim3(BN), dim3(128), 0, stream>>>(qf, kxf, posH, idx21, bigA);
    gemm(bigA, g1h, g1_b, nullptr, a1H, nullptr, nullptr, (int)BN16, 128, 128, 1);
    gemm(a1H, g2h, g2_b, nullptr, bigA, nullptr, nullptr, (int)BN16, 128, 128, 0);

    // softmax over neighbors + weighted sum
    attn_kernel<<<dim3(BN), dim3(128), 0, stream>>>(bigA, vxf, posH, idx21, resH);

    // fc2 + residual f3, store transposed (B, C, N) into d_out
    gemm(resH, fc2h, fc2_b, nullptr, nullptr, f3f, out, BN, 128, 128, 0);
}
